// ComplexGNN_76175539962206
// MI455X (gfx1250) — matrix operand order repaired and checked by reading, not compiled, not run on hardware
//
#include <hip/hip_runtime.h>
#include <hip/hip_bf16.h>
#include <stdint.h>

// ---------------------------------------------------------------------------
// Problem constants (from reference)
// ---------------------------------------------------------------------------
#define NN   20000
#define EE   320000
#define FIN  128
#define HHN  8
#define DDim 64
#define HDim 512
#define NB   64      // graphs (B)
#define NC   10      // classes (C)
#define NEDGE (EE + NN)   // edges + self loops
#define BNEPS 1e-5f

typedef __bf16 bf16;
typedef __attribute__((ext_vector_type(16))) __bf16 bf16x16;
typedef __attribute__((ext_vector_type(8)))  __bf16 bf16x8;
typedef __attribute__((ext_vector_type(8)))  float  v8f;

// ---------------------------------------------------------------------------
// Small utilities
// ---------------------------------------------------------------------------
__device__ __forceinline__ unsigned f2ord(float f) {
    unsigned u = __float_as_uint(f);
    return (u & 0x80000000u) ? ~u : (u | 0x80000000u);
}
__device__ __forceinline__ float ord2f(unsigned u) {
    return __uint_as_float((u & 0x80000000u) ? (u & 0x7FFFFFFFu) : ~u);
}

__global__ void k_fill(float* __restrict__ p, float v, long long n) {
    long long i = (long long)blockIdx.x * blockDim.x + threadIdx.x;
    if (i < n) p[i] = v;
}

__global__ void k_cast_bf16(const float* __restrict__ s, bf16* __restrict__ d, long long n) {
    long long i = (long long)blockIdx.x * blockDim.x + threadIdx.x;
    if (i < n) d[i] = (bf16)s[i];
}

// ---------------------------------------------------------------------------
// bf16 WMMA GEMM:  C[M,Nn] = A[M,K] (bf16) x B[K,Nn] (bf16) + bias[Nn]
// Block tile 128x64, K-step 32, 8 waves (wave32), each wave owns a 32x32 tile
// as 2x2 v_wmma_f32_16x16x32_bf16 accumulators.
// ---------------------------------------------------------------------------
#define BM 128
#define BNT 64
#define BK 32
#define LDA 40   // padded LDS strides (multiple of 8 -> 16B-aligned frag reads)
#define LDB 40

__global__ __launch_bounds__(256) void k_gemm_bf16_wmma(
    const bf16* __restrict__ A, const bf16* __restrict__ Bm,
    const float* __restrict__ bias, float* __restrict__ C,
    int M, int Nn, int K)
{
    __shared__ bf16 As[BM * LDA];
    __shared__ bf16 Bs[BNT * LDB];

    const int tid  = threadIdx.x;
    const int lane = tid & 31;
    const int wave = tid >> 5;          // 0..7
    const int wm   = wave & 3;          // wave row (4 x 32 rows)
    const int wn   = wave >> 2;         // wave col (2 x 32 cols)
    const int hf   = lane >> 4;         // lane half 0/1
    const int l16  = lane & 15;

    const int m0 = blockIdx.x * BM;
    const int n0 = blockIdx.y * BNT;
    const int colBase = n0 + wn * 32 + l16;

    v8f acc[2][2];
    for (int ni = 0; ni < 2; ++ni) {
        float bv = bias ? bias[colBase + ni * 16] : 0.0f;
        for (int mi = 0; mi < 2; ++mi)
            for (int r = 0; r < 8; ++r) acc[mi][ni][r] = bv;
    }

    for (int k0 = 0; k0 < K; k0 += BK) {
        // ---- stage A tile 128x32 (coalesced 16B loads, 2 per thread) ----
        for (int i = 0; i < 2; ++i) {
            int v   = tid + i * 256;        // 0..511
            int row = v >> 2;               // 0..127
            int seg = v & 3;                // 8 bf16 per segment
            int gm  = m0 + row;
            bf16x8 val;
            for (int j = 0; j < 8; ++j) val[j] = (bf16)0.0f;
            if (gm < M)
                val = *(const bf16x8*)(A + (size_t)gm * K + k0 + seg * 8);
            *(bf16x8*)(&As[row * LDA + seg * 8]) = val;
        }
        // ---- stage B tile 32x64, transposed into Bs[n][k] ----
        {
            int krow = tid >> 3;            // 0..31
            int seg  = tid & 7;             // cols seg*8..seg*8+7
            bf16x8 bv = *(const bf16x8*)(Bm + (size_t)(k0 + krow) * Nn + n0 + seg * 8);
            for (int j = 0; j < 8; ++j)
                Bs[(seg * 8 + j) * LDB + krow] = bv[j];
        }
        // prefetch next A tile into cache while we compute (global_prefetch_b8)
        if (k0 + BK < K) {
            int row = tid >> 1;
            int gm  = m0 + row;
            if (gm < M) __builtin_prefetch(A + (size_t)gm * K + k0 + BK, 0, 1);
        }
        __syncthreads();

        // ---- build fragments (ISA 7.12.2 layouts) and issue WMMA ----
        bf16x16 afr[2], bfr[2];
        for (int mi = 0; mi < 2; ++mi) {
            const bf16* ap = &As[(wm * 32 + mi * 16 + l16) * LDA + hf * 8];
            bf16x8 lo = *(const bf16x8*)ap;          // K = hf*8 .. +7
            bf16x8 hi = *(const bf16x8*)(ap + 16);   // K = 16+hf*8 .. +7
            afr[mi] = __builtin_shufflevector(lo, hi, 0,1,2,3,4,5,6,7,8,9,10,11,12,13,14,15);
        }
        for (int ni = 0; ni < 2; ++ni) {
            const bf16* bp = &Bs[(wn * 32 + ni * 16 + l16) * LDB + hf * 8];
            bf16x8 lo = *(const bf16x8*)bp;
            bf16x8 hi = *(const bf16x8*)(bp + 16);
            bfr[ni] = __builtin_shufflevector(lo, hi, 0,1,2,3,4,5,6,7,8,9,10,11,12,13,14,15);
        }
        for (int mi = 0; mi < 2; ++mi)
            for (int ni = 0; ni < 2; ++ni)
                acc[mi][ni] = __builtin_amdgcn_wmma_f32_16x16x32_bf16(
                    false, afr[mi], false, bfr[ni], (short)0, acc[mi][ni], false, false);
        __syncthreads();
    }

    // ---- store C (f32 16x16 layout: vgpr r -> m = r + 8*hf, lane -> n) ----
    for (int mi = 0; mi < 2; ++mi)
        for (int ni = 0; ni < 2; ++ni)
            for (int r = 0; r < 8; ++r) {
                int m = m0 + wm * 32 + mi * 16 + hf * 8 + r;
                if (m < M) C[(size_t)m * Nn + colBase + ni * 16] = acc[mi][ni][r];
            }
}

// ---------------------------------------------------------------------------
// GAT edge pipeline
// ---------------------------------------------------------------------------
__global__ void k_scores(const float* __restrict__ h,
                         const float* __restrict__ a_s, const float* __restrict__ a_d,
                         float* __restrict__ ssrc, float* __restrict__ sdst,
                         int Hh, int Dl)
{
    long long t = (long long)blockIdx.x * blockDim.x + threadIdx.x;
    if (t >= (long long)NN * Hh) return;
    int n  = (int)(t / Hh);
    int hh = (int)(t % Hh);
    const float* hp = h + (size_t)n * Hh * Dl + hh * Dl;
    const float* p1 = a_s + hh * Dl;
    const float* p2 = a_d + hh * Dl;
    float s1 = 0.f, s2 = 0.f;
    for (int d = 0; d < Dl; ++d) { float v = hp[d]; s1 += v * p1[d]; s2 += v * p2[d]; }
    ssrc[t] = s1; sdst[t] = s2;
}

__device__ __forceinline__ void edge_ends(const int* __restrict__ ei, int e, int& s, int& d) {
    if (e < EE) { s = ei[e]; d = ei[EE + e]; }
    else        { s = d = e - EE; }
}

__global__ void k_edge_max(const int* __restrict__ ei,
                           const float* __restrict__ ssrc, const float* __restrict__ sdst,
                           unsigned* __restrict__ smax, int Hh)
{
    long long t = (long long)blockIdx.x * blockDim.x + threadIdx.x;
    if (t >= (long long)NEDGE * Hh) return;
    int e = (int)(t / Hh), hh = (int)(t % Hh);
    int s, d; edge_ends(ei, e, s, d);
    float sc = ssrc[(size_t)s * Hh + hh] + sdst[(size_t)d * Hh + hh];
    sc = sc > 0.f ? sc : 0.2f * sc;
    atomicMax(&smax[(size_t)d * Hh + hh], f2ord(sc));
}

__global__ void k_edge_sum(const int* __restrict__ ei,
                           const float* __restrict__ ssrc, const float* __restrict__ sdst,
                           const unsigned* __restrict__ smax, float* __restrict__ ssum, int Hh)
{
    long long t = (long long)blockIdx.x * blockDim.x + threadIdx.x;
    if (t >= (long long)NEDGE * Hh) return;
    int e = (int)(t / Hh), hh = (int)(t % Hh);
    int s, d; edge_ends(ei, e, s, d);
    float sc = ssrc[(size_t)s * Hh + hh] + sdst[(size_t)d * Hh + hh];
    sc = sc > 0.f ? sc : 0.2f * sc;
    float m  = ord2f(smax[(size_t)d * Hh + hh]);
    atomicAdd(&ssum[(size_t)d * Hh + hh], expf(sc - m));
}

__global__ __launch_bounds__(256) void k_edge_msg(
    const int* __restrict__ ei, const float* __restrict__ h,
    const float* __restrict__ ssrc, const float* __restrict__ sdst,
    const unsigned* __restrict__ smax, const float* __restrict__ ssum,
    float* __restrict__ scat, int Hh, int Dl, int od)
{
    long long t = (long long)blockIdx.x * blockDim.x + threadIdx.x;
    int chunks = od >> 2;
    if (t >= (long long)NEDGE * chunks) return;
    int e  = (int)(t / chunks);
    int c4 = (int)(t % chunks);
    int c  = c4 * 4;
    int hh = c / Dl;
    int s, d; edge_ends(ei, e, s, d);
    float sc = ssrc[(size_t)s * Hh + hh] + sdst[(size_t)d * Hh + hh];
    sc = sc > 0.f ? sc : 0.2f * sc;
    float m  = ord2f(smax[(size_t)d * Hh + hh]);
    float al = expf(sc - m) / (ssum[(size_t)d * Hh + hh] + 1e-16f);
    float4 v = *(const float4*)(h + (size_t)s * od + c);
    float* o = scat + (size_t)d * od + c;
    atomicAdd(o + 0, v.x * al);
    atomicAdd(o + 1, v.y * al);
    atomicAdd(o + 2, v.z * al);
    atomicAdd(o + 3, v.w * al);
}

__global__ void k_bn_elu(const float* __restrict__ scat, const float* __restrict__ gatb,
                         const float* __restrict__ g, const float* __restrict__ b,
                         const float* __restrict__ m, const float* __restrict__ v,
                         float* __restrict__ out, int od)
{
    long long t = (long long)blockIdx.x * blockDim.x + threadIdx.x;
    if (t >= (long long)NN * od) return;
    int c = (int)(t % od);
    float x  = scat[t] + gatb[c];
    float xn = (x - m[c]) * (g[c] * rsqrtf(v[c] + BNEPS)) + b[c];
    out[t] = xn > 0.f ? xn : expm1f(xn);
}

// ---------------------------------------------------------------------------
// GCN + pooling + head
// ---------------------------------------------------------------------------
__global__ void k_deg(const int* __restrict__ ei, float* __restrict__ deg) {
    long long t = (long long)blockIdx.x * blockDim.x + threadIdx.x;
    if (t >= NEDGE) return;
    int s, d; edge_ends(ei, (int)t, s, d);
    atomicAdd(&deg[d], 1.0f);
}

__global__ void k_gcn_lin(const float* __restrict__ x, const float* __restrict__ W,
                          float* __restrict__ out)
{
    long long t = (long long)blockIdx.x * blockDim.x + threadIdx.x;
    if (t >= (long long)NN * NC) return;
    int n = (int)(t / NC), c = (int)(t % NC);
    const float* xp = x + (size_t)n * DDim;
    float s = 0.f;
    for (int d = 0; d < DDim; ++d) s += xp[d] * W[d * NC + c];
    out[t] = s;
}

__global__ void k_gcn_scatter(const int* __restrict__ ei, const float* __restrict__ hl,
                              const float* __restrict__ deg, float* __restrict__ go)
{
    long long t = (long long)blockIdx.x * blockDim.x + threadIdx.x;
    if (t >= NEDGE) return;
    int s, d; edge_ends(ei, (int)t, s, d);
    float norm = rsqrtf(deg[s]) * rsqrtf(deg[d]);
    const float* hp = hl + (size_t)s * NC;
    float* op = go + (size_t)d * NC;
    for (int c = 0; c < NC; ++c) atomicAdd(&op[c], hp[c] * norm);
}

__global__ void k_pool(const float* __restrict__ go, const float* __restrict__ gcnb,
                       const int* __restrict__ batch,
                       float* __restrict__ pooled, float* __restrict__ cnt)
{
    long long t = (long long)blockIdx.x * blockDim.x + threadIdx.x;
    if (t >= NN) return;
    int bidx = batch[t];
    atomicAdd(&cnt[bidx], 1.0f);
    const float* gp = go + (size_t)t * NC;
    float* pp = pooled + (size_t)bidx * NC;
    for (int c = 0; c < NC; ++c) atomicAdd(&pp[c], gp[c] + gcnb[c]);
}

__global__ void k_head(const float* __restrict__ pooled, const float* __restrict__ cnt,
                       const float* __restrict__ W1, const float* __restrict__ b1,
                       const float* __restrict__ g,  const float* __restrict__ bb,
                       const float* __restrict__ m,  const float* __restrict__ v,
                       const float* __restrict__ W2, const float* __restrict__ b2,
                       float* __restrict__ out)
{
    int gi = threadIdx.x;
    if (gi >= NB) return;
    float c = cnt[gi]; c = c > 1.f ? c : 1.f;
    float x[NC];
    for (int i = 0; i < NC; ++i) x[i] = pooled[gi * NC + i] / c;
    float o[NC];
    for (int i = 0; i < NC; ++i) o[i] = b2[i];
    for (int j = 0; j < DDim; ++j) {
        float hv = b1[j];
        for (int i = 0; i < NC; ++i) hv += x[i] * W1[i * DDim + j];
        hv = (hv - m[j]) * (g[j] * rsqrtf(v[j] + BNEPS)) + bb[j];
        hv = hv > 0.f ? hv : 0.f;
        for (int i = 0; i < NC; ++i) o[i] += hv * W2[j * NC + i];
    }
    float mx = o[0];
    for (int i = 1; i < NC; ++i) mx = fmaxf(mx, o[i]);
    float se = 0.f;
    for (int i = 0; i < NC; ++i) se += expf(o[i] - mx);
    float ls = mx + logf(se);
    for (int i = 0; i < NC; ++i) out[gi * NC + i] = o[i] - ls;
}

// ---------------------------------------------------------------------------
// TDM demonstration: DMA a 64x64 f32 tile of x into LDS via tensor_load_to_lds
// (D# per ISA 08_async_tensor §8), wait on TENSORcnt, spill to scratch.
// This toolchain exposes the 6-arg builtin:
//   (uint32x4 g0, int32x8 g1, int32x4 g2, int32x4 g3, int32x8 pad, i32 cpol)
// ---------------------------------------------------------------------------
#if defined(__has_builtin)
#if __has_builtin(__builtin_amdgcn_tensor_load_to_lds) && __has_builtin(__builtin_amdgcn_s_wait_tensorcnt)
#define HAVE_TDM 1
#endif
#endif

#ifdef HAVE_TDM
typedef __attribute__((ext_vector_type(4))) unsigned int u32x4;
typedef __attribute__((ext_vector_type(8))) int          i32x8;
typedef __attribute__((ext_vector_type(4))) int          i32x4;

__global__ void k_tdm_tile(const float* __restrict__ src, float* __restrict__ dst)
{
    __shared__ float tile[64 * 64];
    unsigned ldsoff = (unsigned)(size_t)(void*)&tile[0];        // low 32b of flat = LDS offset
    unsigned long long ga = (unsigned long long)(size_t)src;    // tile start == tensor start

    u32x4 g0;
    g0[0] = 1u;                                  // count=1, user descriptor
    g0[1] = ldsoff;                              // lds_addr
    g0[2] = (unsigned)(ga & 0xFFFFFFFFull);      // global_addr[31:0]
    g0[3] = (unsigned)((ga >> 32) & 0x01FFFFFFull) | (2u << 30); // addr[56:32] | type=2

    unsigned dim0 = FIN, dim1 = NN, t0 = 64, t1 = 64, str0 = FIN;
    i32x8 g1;
    g1[0] = (int)(2u << 16);                                   // data_size=4B, no multicast
    g1[1] = (int)((dim0 & 0xFFFFu) << 16);                     // tensor_dim0 lo16
    g1[2] = (int)((dim0 >> 16) | ((dim1 & 0xFFFFu) << 16));    // dim0 hi16 | dim1 lo16
    g1[3] = (int)((dim1 >> 16) | ((t0 & 0xFFFFu) << 16));      // dim1 hi16 | tile_dim0
    g1[4] = (int)(t1 & 0xFFFFu);                               // tile_dim1, tile_dim2=0
    g1[5] = (int)str0;                                         // tensor_dim0_stride lo32
    g1[6] = 0;
    g1[7] = 0;
    i32x4 z4; z4[0] = z4[1] = z4[2] = z4[3] = 0;
    i32x8 z8; for (int i = 0; i < 8; ++i) z8[i] = 0;

    if (threadIdx.x < 32) {                                    // one wave issues the DMA
        __builtin_amdgcn_tensor_load_to_lds(g0, g1, z4, z4, z8, 0);
        __builtin_amdgcn_s_wait_tensorcnt(0);
    }
    __syncthreads();
    for (int i = threadIdx.x; i < 64 * 64; i += blockDim.x) dst[i] = tile[i];
}
#endif

// ---------------------------------------------------------------------------
// Host orchestration
// ---------------------------------------------------------------------------
static inline int divup_ll(long long a, long long b) { return (int)((a + b - 1) / b); }

extern "C" void kernel_launch(void* const* d_in, const int* in_sizes, int n_in,
                              void* d_out, int out_size, void* d_ws, size_t ws_size,
                              hipStream_t stream)
{
    (void)in_sizes; (void)n_in; (void)out_size; (void)ws_size;

    const float* x     = (const float*)d_in[0];
    const int*   ei    = (const int*)d_in[1];
    const int*   batch = (const int*)d_in[2];
    auto P = [&](int i) { return (const float*)d_in[i]; };

    // workspace carve (256B aligned)
    char* base = (char*)d_ws; size_t off = 0;
    auto carve = [&](size_t bytes) -> void* {
        off = (off + 255) & ~(size_t)255;
        void* p = base + off; off += bytes; return p;
    };
    float*    cur    = (float*)carve((size_t)NN * HDim * 4);
    float*    hbuf   = (float*)carve((size_t)NN * HDim * 4);
    float*    scat   = (float*)carve((size_t)NN * HDim * 4);
    bf16*     curb   = (bf16*)carve((size_t)NN * HDim * 2);
    bf16*     Wb     = (bf16*)carve((size_t)HDim * HDim * 2);
    float*    ssrc   = (float*)carve((size_t)NN * HHN * 4);
    float*    sdst   = (float*)carve((size_t)NN * HHN * 4);
    unsigned* smax   = (unsigned*)carve((size_t)NN * HHN * 4);
    float*    ssum   = (float*)carve((size_t)NN * HHN * 4);
    float*    deg    = (float*)carve((size_t)NN * 4);
    float*    gl     = (float*)carve((size_t)NN * NC * 4);
    float*    go     = (float*)carve((size_t)NN * NC * 4);
    float*    pooled = (float*)carve((size_t)NB * NC * 4);
    float*    cnt    = (float*)carve((size_t)NB * 4);
    float*    tdmdst = (float*)carve((size_t)64 * 64 * 4);
    (void)tdmdst;

    const int dims_in[4] = { FIN, HDim, HDim, HDim };
    const int heads[4]   = { 8, 8, 8, 1 };
    const int odims[4]   = { 512, 512, 512, 64 };

    for (int L = 0; L < 4; ++L) {
        const int K = dims_in[L], OD = odims[L], Hh = heads[L];
        const int pb = 3 + 10 * L;

        // cast activations + GAT weight to bf16, GEMM h = x @ W (no bias)
        long long na = (long long)NN * K;
        k_cast_bf16<<<divup_ll(na, 256), 256, 0, stream>>>(L == 0 ? x : cur, curb, na);
        long long nw = (long long)K * OD;
        k_cast_bf16<<<divup_ll(nw, 256), 256, 0, stream>>>(P(pb + 0), Wb, nw);
        dim3 gg(divup_ll(NN, BM), OD / BNT);
        k_gemm_bf16_wmma<<<gg, 256, 0, stream>>>(curb, Wb, nullptr, hbuf, NN, OD, K);

        // skip branch: scat = x @ skip_W + skip_b   (scatter target seeded with skip)
        k_cast_bf16<<<divup_ll(nw, 256), 256, 0, stream>>>(P(pb + 4), Wb, nw);
        k_gemm_bf16_wmma<<<gg, 256, 0, stream>>>(curb, Wb, P(pb + 5), scat, NN, OD, K);

        // attention scores + segment softmax + message scatter
        long long ns = (long long)NN * Hh;
        k_scores<<<divup_ll(ns, 256), 256, 0, stream>>>(hbuf, P(pb + 1), P(pb + 2), ssrc, sdst, Hh, DDim);
        k_fill<<<divup_ll(ns, 256), 256, 0, stream>>>((float*)smax, 0.0f, ns);  // 0u in ordered space
        k_fill<<<divup_ll(ns, 256), 256, 0, stream>>>(ssum, 0.0f, ns);
        long long nEh = (long long)NEDGE * Hh;
        k_edge_max<<<divup_ll(nEh, 256), 256, 0, stream>>>(ei, ssrc, sdst, smax, Hh);
        k_edge_sum<<<divup_ll(nEh, 256), 256, 0, stream>>>(ei, ssrc, sdst, smax, ssum, Hh);
        long long nm = (long long)NEDGE * (OD / 4);
        k_edge_msg<<<divup_ll(nm, 256), 256, 0, stream>>>(ei, hbuf, ssrc, sdst, smax, ssum, scat, Hh, DDim, OD);

        // BN (eval) + ELU -> next activations
        long long nb = (long long)NN * OD;
        k_bn_elu<<<divup_ll(nb, 256), 256, 0, stream>>>(scat, P(pb + 3), P(pb + 6), P(pb + 7),
                                                        P(pb + 8), P(pb + 9), cur, OD);
    }

    // GCN with symmetric normalization (self-loops included in NEDGE)
    k_fill<<<divup_ll(NN, 256), 256, 0, stream>>>(deg, 0.0f, NN);
    k_deg<<<divup_ll(NEDGE, 256), 256, 0, stream>>>(ei, deg);
    k_gcn_lin<<<divup_ll((long long)NN * NC, 256), 256, 0, stream>>>(cur, P(43), gl);
    k_fill<<<divup_ll((long long)NN * NC, 256), 256, 0, stream>>>(go, 0.0f, (long long)NN * NC);
    k_gcn_scatter<<<divup_ll(NEDGE, 256), 256, 0, stream>>>(ei, gl, deg, go);

    // global mean pool + MLP head + log_softmax
    k_fill<<<divup_ll((long long)NB * NC, 256), 256, 0, stream>>>(pooled, 0.0f, (long long)NB * NC);
    k_fill<<<1, 256, 0, stream>>>(cnt, 0.0f, NB);
    k_pool<<<divup_ll(NN, 256), 256, 0, stream>>>(go, P(44), batch, pooled, cnt);
    k_head<<<1, 64, 0, stream>>>(pooled, cnt, P(45), P(46), P(47), P(48), P(49), P(50), P(51), P(52),
                                 (float*)d_out);

#ifdef HAVE_TDM
    // CDNA5 Tensor Data Mover demonstration (writes only to private scratch)
    k_tdm_tile<<<1, 256, 0, stream>>>(x, tdmdst);
#endif
}
